// MultiHeadAttention_11819749999197
// MI455X (gfx1250) — matrix operand order repaired and checked by reading, not compiled, not run on hardware
//
#include <hip/hip_runtime.h>
#include <hip/hip_bf16.h>

// MI455X / gfx1250, wave32. bf16 WMMA (f32 accum) for all GEMMs.

typedef __bf16 bf16;
typedef __attribute__((ext_vector_type(16))) __bf16 v16bf;
typedef __attribute__((ext_vector_type(8)))  float  v8f;

#define B_  4
#define T_  2048
#define D_  1024
#define H_  16
#define HD_ 64

union FragAB { v16bf v; uint4 u[2]; bf16 e[16]; };
union Pack8  { bf16 e[8]; uint4 u; };

static __device__ __forceinline__ v8f wmma_bf16(const v16bf& a, const v16bf& b, const v8f& c) {
  // (neg_a, A, neg_b, B, c_mod, C, reuse_a, reuse_b)
  return __builtin_amdgcn_wmma_f32_16x16x32_bf16(false, a, false, b, (short)0, c, false, false);
}

// ---------------- Kernel 0: weights fp32 -> bf16 (one-time, stays in L2) ---
__global__ __launch_bounds__(256) void k_wcvt(const float* __restrict__ wq,
                                              const float* __restrict__ wk,
                                              const float* __restrict__ wv,
                                              const float* __restrict__ wo,
                                              bf16* __restrict__ out) {
  int idx = blockIdx.x * 256 + threadIdx.x;   // 1M threads, 4 floats each
  int m   = idx >> 18;                        // / 262144 -> matrix id
  int off = idx & 262143;
  const float* src = (m == 0) ? wq : (m == 1) ? wk : (m == 2) ? wv : wo;
  float4 f = reinterpret_cast<const float4*>(src)[off];
  union { bf16 e[4]; uint2 u; } p;
  p.e[0] = (bf16)f.x; p.e[1] = (bf16)f.y; p.e[2] = (bf16)f.z; p.e[3] = (bf16)f.w;
  *reinterpret_cast<uint2*>(out + (size_t)m * (D_ * D_) + (size_t)off * 4) = p.u;
}

// ---------------- Kernel 1: fused QKV projection ---------------------------
// Q,K: computed transposed (A=W rows M=e, B=x^T) -> packed row-major stores.
// V:   computed direct (A=x, B=W) -> packed stores into Vt[b,h,hd,t].
#define LDSROW 1032  // 1024 + 8 bf16 pad (16B) to rotate LDS banks per row

__global__ __launch_bounds__(256) void k_qkv(const float* __restrict__ x,
                                             const bf16* __restrict__ Wb,
                                             bf16* __restrict__ Q,
                                             bf16* __restrict__ K,
                                             bf16* __restrict__ Vt) {
  __shared__ bf16 xs[16 * LDSROW];
  const int rt   = blockIdx.x;            // 16-row tile over B*T (512 tiles)
  const int tid  = threadIdx.x;
  const long g0  = (long)rt * 16;         // global row base in [0, B*T)

  // Stage x tile (16 x 1024 fp32) -> bf16 LDS
  for (int i = tid; i < 16 * 256; i += 256) {
    int r = i >> 8, c4 = i & 255;
    float4 f = reinterpret_cast<const float4*>(x + (size_t)(g0 + r) * D_)[c4];
    union { bf16 e[4]; uint2 u; } p;
    p.e[0] = (bf16)f.x; p.e[1] = (bf16)f.y; p.e[2] = (bf16)f.z; p.e[3] = (bf16)f.w;
    *reinterpret_cast<uint2*>(&xs[r * LDSROW + c4 * 4]) = p.u;
  }
  __syncthreads();

  const int lane = tid & 31, wave = tid >> 5;
  const int ln15 = lane & 15, hi = lane >> 4;
  const int o8 = hi * 8, o16 = hi * 16;
  const bf16* xrow = &xs[ln15 * LDSROW];

  for (int j = wave; j < 48; j += 8) {          // 3 proj x 16 col-groups(64)
    const int p  = j >> 4;                       // 0:Q 1:K 2:V
    const int cg = j & 15;
    const int ebase = cg * 64;
    const bf16* W = Wb + (size_t)p * (D_ * D_);
    v8f acc[4] = {{}, {}, {}, {}};

    if (p < 2) {
      // Q/K transposed: A = W e-rows (M=e), B = x^T (N=t, K-dim=d)
      const bf16* wr0 = W + (size_t)(ebase +  0 + ln15) * D_;
      const bf16* wr1 = W + (size_t)(ebase + 16 + ln15) * D_;
      const bf16* wr2 = W + (size_t)(ebase + 32 + ln15) * D_;
      const bf16* wr3 = W + (size_t)(ebase + 48 + ln15) * D_;
      for (int d0 = 0; d0 < D_; d0 += 32) {
        FragAB bx;
        bx.u[0] = *reinterpret_cast<const uint4*>(xrow + d0 + o8);
        bx.u[1] = *reinterpret_cast<const uint4*>(xrow + d0 + o8 + 16);
        FragAB a0, a1, a2, a3;
        a0.u[0] = *reinterpret_cast<const uint4*>(wr0 + d0 + o8);
        a0.u[1] = *reinterpret_cast<const uint4*>(wr0 + d0 + o8 + 16);
        a1.u[0] = *reinterpret_cast<const uint4*>(wr1 + d0 + o8);
        a1.u[1] = *reinterpret_cast<const uint4*>(wr1 + d0 + o8 + 16);
        a2.u[0] = *reinterpret_cast<const uint4*>(wr2 + d0 + o8);
        a2.u[1] = *reinterpret_cast<const uint4*>(wr2 + d0 + o8 + 16);
        a3.u[0] = *reinterpret_cast<const uint4*>(wr3 + d0 + o8);
        a3.u[1] = *reinterpret_cast<const uint4*>(wr3 + d0 + o8 + 16);
        acc[0] = wmma_bf16(a0.v, bx.v, acc[0]);
        acc[1] = wmma_bf16(a1.v, bx.v, acc[1]);
        acc[2] = wmma_bf16(a2.v, bx.v, acc[2]);
        acc[3] = wmma_bf16(a3.v, bx.v, acc[3]);
      }
      // D: reg r -> e = ebase + tt*16 + o8 + r ; lane -> t = g0 + ln15
      bf16* dst = ((p == 0) ? Q : K) + (size_t)(g0 + ln15) * D_ + ebase + o8;
#pragma unroll
      for (int tt = 0; tt < 4; ++tt) {
        Pack8 pk;
#pragma unroll
        for (int r = 0; r < 8; ++r) pk.e[r] = (bf16)acc[tt][r];
        *reinterpret_cast<uint4*>(dst + tt * 16) = pk.u;
      }
    } else {
      // V direct: A = x (M=t), B = W^T cols (N=e)
      const bf16* wr0 = W + (size_t)(ebase +  0 + ln15) * D_;
      const bf16* wr1 = W + (size_t)(ebase + 16 + ln15) * D_;
      const bf16* wr2 = W + (size_t)(ebase + 32 + ln15) * D_;
      const bf16* wr3 = W + (size_t)(ebase + 48 + ln15) * D_;
      for (int d0 = 0; d0 < D_; d0 += 32) {
        FragAB ax;
        ax.u[0] = *reinterpret_cast<const uint4*>(xrow + d0 + o16);
        ax.u[1] = *reinterpret_cast<const uint4*>(xrow + d0 + o16 + 8);
        FragAB b0, b1, b2, b3;
        b0.u[0] = *reinterpret_cast<const uint4*>(wr0 + d0 + o16);
        b0.u[1] = *reinterpret_cast<const uint4*>(wr0 + d0 + o16 + 8);
        b1.u[0] = *reinterpret_cast<const uint4*>(wr1 + d0 + o16);
        b1.u[1] = *reinterpret_cast<const uint4*>(wr1 + d0 + o16 + 8);
        b2.u[0] = *reinterpret_cast<const uint4*>(wr2 + d0 + o16);
        b2.u[1] = *reinterpret_cast<const uint4*>(wr2 + d0 + o16 + 8);
        b3.u[0] = *reinterpret_cast<const uint4*>(wr3 + d0 + o16);
        b3.u[1] = *reinterpret_cast<const uint4*>(wr3 + d0 + o16 + 8);
        acc[0] = wmma_bf16(ax.v, b0.v, acc[0]);
        acc[1] = wmma_bf16(ax.v, b1.v, acc[1]);
        acc[2] = wmma_bf16(ax.v, b2.v, acc[2]);
        acc[3] = wmma_bf16(ax.v, b3.v, acc[3]);
      }
      // D: reg r -> t = g0%T + o8 + r ; lane -> e = ebase + tt*16 + ln15
      const long b_ = g0 / T_;
      const long tb = (g0 % T_) + o8;
#pragma unroll
      for (int tt = 0; tt < 4; ++tt) {
        const int e = ebase + tt * 16 + ln15;
        const int h = e >> 6, hd = e & 63;
        Pack8 pk;
#pragma unroll
        for (int r = 0; r < 8; ++r) pk.e[r] = (bf16)acc[tt][r];
        *reinterpret_cast<uint4*>(Vt + (((size_t)(b_ * H_ + h) * HD_ + hd) * T_ + tb)) = pk.u;
      }
    }
  }
}

// ---------------- Kernel 2: causal flash attention -------------------------
// One wave per 16-query tile. S^T = K·Q^T, online softmax, O^T = V^T·P^T.
__global__ __launch_bounds__(256) void k_attn(const bf16* __restrict__ Q,
                                              const bf16* __restrict__ K,
                                              const bf16* __restrict__ Vt,
                                              bf16* __restrict__ O) {
  const int bh = blockIdx.x;                 // b*16 + h
  const int b  = bh >> 4, h = bh & 15;
  const int wave = threadIdx.x >> 5, lane = threadIdx.x & 31;
  const int qt = blockIdx.y * 8 + wave;      // 0..127
  const int ln15 = lane & 15, hi = lane >> 4;
  const int o8 = hi * 8, o16 = hi * 16;

  // Q^T B-fragments (N = query = ln15, K-dim = hd)
  const size_t rowQ = ((size_t)b * T_ + qt * 16 + ln15) * D_ + h * HD_;
  FragAB qb0, qb1;
  qb0.u[0] = *reinterpret_cast<const uint4*>(Q + rowQ + o8);
  qb0.u[1] = *reinterpret_cast<const uint4*>(Q + rowQ + o8 + 16);
  qb1.u[0] = *reinterpret_cast<const uint4*>(Q + rowQ + 32 + o8);
  qb1.u[1] = *reinterpret_cast<const uint4*>(Q + rowQ + 32 + o8 + 16);

  const float scale = 0.125f;                // 1/sqrt(64)
  float m = -__builtin_inff(), l = 0.f;
  v8f o0 = {}, o1 = {}, o2 = {}, o3 = {};

  const int q    = qt * 16 + ln15;           // this lane's query column
  const int nkb  = ((qt * 16 + 15) >> 5) + 1;
  const bf16* Kb = K + (size_t)b * T_ * D_ + h * HD_;
  const bf16* Vb = Vt + (size_t)(b * H_ + h) * HD_ * T_;

  for (int kb = 0; kb < nkb; ++kb) {
    const int kbase = kb * 32;
    // S^T tiles: A = K rows (M = key), B = Q^T
    v8f st0 = {}, st1 = {};
    const bf16* krow = Kb + (size_t)(kbase + ln15) * D_;
    if (kb + 1 < nkb) __builtin_prefetch(krow + 32 * (size_t)D_, 0, 0);
    FragAB a0, a1;
    a0.u[0] = *reinterpret_cast<const uint4*>(krow + o8);
    a0.u[1] = *reinterpret_cast<const uint4*>(krow + o8 + 16);
    a1.u[0] = *reinterpret_cast<const uint4*>(krow + 32 + o8);
    a1.u[1] = *reinterpret_cast<const uint4*>(krow + 32 + o8 + 16);
    st0 = wmma_bf16(a0.v, qb0.v, st0);
    st0 = wmma_bf16(a1.v, qb1.v, st0);
    const bf16* krow2 = krow + 16 * (size_t)D_;
    a0.u[0] = *reinterpret_cast<const uint4*>(krow2 + o8);
    a0.u[1] = *reinterpret_cast<const uint4*>(krow2 + o8 + 16);
    a1.u[0] = *reinterpret_cast<const uint4*>(krow2 + 32 + o8);
    a1.u[1] = *reinterpret_cast<const uint4*>(krow2 + 32 + o8 + 16);
    st1 = wmma_bf16(a0.v, qb0.v, st1);
    st1 = wmma_bf16(a1.v, qb1.v, st1);

    // scale + causal mask (key = kbase + {0,16} + o8 + r)
    float s0[8], s1[8];
#pragma unroll
    for (int r = 0; r < 8; ++r) {
      int k0 = kbase + o8 + r;
      s0[r] = (k0      <= q) ? st0[r] * scale : -__builtin_inff();
      s1[r] = (k0 + 16 <= q) ? st1[r] * scale : -__builtin_inff();
    }
    // online softmax: lane + partner(lane^16) together cover 32 keys of query q
    float bm = s0[0];
#pragma unroll
    for (int r = 1; r < 8; ++r) bm = fmaxf(bm, s0[r]);
#pragma unroll
    for (int r = 0; r < 8; ++r) bm = fmaxf(bm, s1[r]);
    bm = fmaxf(bm, __shfl_xor(bm, 16, 32));
    const float mnew = fmaxf(m, bm);
    const float resc = __expf(m - mnew);
    o0 = o0 * resc; o1 = o1 * resc; o2 = o2 * resc; o3 = o3 * resc;

    float e0[8], e1[8], ls = 0.f;
#pragma unroll
    for (int r = 0; r < 8; ++r) {
      e0[r] = __expf(s0[r] - mnew);
      e1[r] = __expf(s1[r] - mnew);
      ls += e0[r] + e1[r];
    }
    ls += __shfl_xor(ls, 16, 32);
    l = l * resc + ls;
    m = mnew;

    // P^T B-fragment (N = query, K-dim = key) from this lane's own e0/e1
    FragAB pb;
#pragma unroll
    for (int r = 0; r < 8; ++r) {
      float x0 = __shfl_xor(e0[r], 16, 32);
      float x1 = __shfl_xor(e1[r], 16, 32);
      pb.e[r]     = (bf16)e0[r];   // keys 0..7   | 8..15
      pb.e[r + 8] = (bf16)e1[r];   // keys 16..23 | 24..31
    }
    // O^T += V^T · P^T   (A = V^T tile: M = hd, K-dim = key)
    const bf16* vrow = Vb + (size_t)ln15 * T_ + kbase + o8;
    FragAB va;
    va.u[0] = *reinterpret_cast<const uint4*>(vrow);
    va.u[1] = *reinterpret_cast<const uint4*>(vrow + 16);
    o0 = wmma_bf16(va.v, pb.v, o0);
    vrow += 16 * (size_t)T_;
    va.u[0] = *reinterpret_cast<const uint4*>(vrow);
    va.u[1] = *reinterpret_cast<const uint4*>(vrow + 16);
    o1 = wmma_bf16(va.v, pb.v, o1);
    vrow += 16 * (size_t)T_;
    va.u[0] = *reinterpret_cast<const uint4*>(vrow);
    va.u[1] = *reinterpret_cast<const uint4*>(vrow + 16);
    o2 = wmma_bf16(va.v, pb.v, o2);
    vrow += 16 * (size_t)T_;
    va.u[0] = *reinterpret_cast<const uint4*>(vrow);
    va.u[1] = *reinterpret_cast<const uint4*>(vrow + 16);
    o3 = wmma_bf16(va.v, pb.v, o3);
  }

  // normalize & store O (row-major bf16 [B*T, D]); e = h*64 + hm*16 + o8 + r
  const float rl = 1.0f / l;
  bf16* dst = O + ((size_t)b * T_ + qt * 16 + ln15) * D_ + h * HD_ + o8;
  Pack8 pk;
#pragma unroll
  for (int r = 0; r < 8; ++r) pk.e[r] = (bf16)(o0[r] * rl);
  *reinterpret_cast<uint4*>(dst) = pk.u;
#pragma unroll
  for (int r = 0; r < 8; ++r) pk.e[r] = (bf16)(o1[r] * rl);
  *reinterpret_cast<uint4*>(dst + 16) = pk.u;
#pragma unroll
  for (int r = 0; r < 8; ++r) pk.e[r] = (bf16)(o2[r] * rl);
  *reinterpret_cast<uint4*>(dst + 32) = pk.u;
#pragma unroll
  for (int r = 0; r < 8; ++r) pk.e[r] = (bf16)(o3[r] * rl);
  *reinterpret_cast<uint4*>(dst + 48) = pk.u;
}

// ---------------- Kernel 3: output projection  out = O·Wo^T + bo (fp32) ----
// Transposed compute: A = Wo e-rows (M=e), B = O rows (N=t) -> packed stores.
__global__ __launch_bounds__(256) void k_oproj(const bf16* __restrict__ O,
                                               const bf16* __restrict__ Wo,
                                               const float* __restrict__ bo,
                                               float* __restrict__ out) {
  const int rt = blockIdx.x;                       // 512 row tiles
  const int wave = threadIdx.x >> 5, lane = threadIdx.x & 31;
  const int cg = blockIdx.y * 8 + wave;            // 16 col-groups of 64
  const int ln15 = lane & 15, hi = lane >> 4;
  const int o8 = hi * 8, o16 = hi * 16;
  const int ebase = cg * 64;

  const bf16* orow = O + (size_t)(rt * 16 + ln15) * D_;
  const bf16* wr0 = Wo + (size_t)(ebase +  0 + ln15) * D_;
  const bf16* wr1 = Wo + (size_t)(ebase + 16 + ln15) * D_;
  const bf16* wr2 = Wo + (size_t)(ebase + 32 + ln15) * D_;
  const bf16* wr3 = Wo + (size_t)(ebase + 48 + ln15) * D_;

  v8f acc[4] = {{}, {}, {}, {}};
  for (int d0 = 0; d0 < D_; d0 += 32) {
    FragAB bx;
    bx.u[0] = *reinterpret_cast<const uint4*>(orow + d0 + o8);
    bx.u[1] = *reinterpret_cast<const uint4*>(orow + d0 + o8 + 16);
    FragAB a0, a1, a2, a3;
    a0.u[0] = *reinterpret_cast<const uint4*>(wr0 + d0 + o8);
    a0.u[1] = *reinterpret_cast<const uint4*>(wr0 + d0 + o8 + 16);
    a1.u[0] = *reinterpret_cast<const uint4*>(wr1 + d0 + o8);
    a1.u[1] = *reinterpret_cast<const uint4*>(wr1 + d0 + o8 + 16);
    a2.u[0] = *reinterpret_cast<const uint4*>(wr2 + d0 + o8);
    a2.u[1] = *reinterpret_cast<const uint4*>(wr2 + d0 + o8 + 16);
    a3.u[0] = *reinterpret_cast<const uint4*>(wr3 + d0 + o8);
    a3.u[1] = *reinterpret_cast<const uint4*>(wr3 + d0 + o8 + 16);
    acc[0] = wmma_bf16(a0.v, bx.v, acc[0]);
    acc[1] = wmma_bf16(a1.v, bx.v, acc[1]);
    acc[2] = wmma_bf16(a2.v, bx.v, acc[2]);
    acc[3] = wmma_bf16(a3.v, bx.v, acc[3]);
  }

  // D: reg r -> e = ebase + tt*16 + o8 + r ; lane -> row = rt*16 + ln15
  float* dst = out + (size_t)(rt * 16 + ln15) * D_ + ebase + o8;
#pragma unroll
  for (int tt = 0; tt < 4; ++tt) {
    const float4 bv0 = *reinterpret_cast<const float4*>(bo + ebase + tt * 16 + o8);
    const float4 bv1 = *reinterpret_cast<const float4*>(bo + ebase + tt * 16 + o8 + 4);
    float4 r0, r1;
    r0.x = acc[tt][0] + bv0.x; r0.y = acc[tt][1] + bv0.y;
    r0.z = acc[tt][2] + bv0.z; r0.w = acc[tt][3] + bv0.w;
    r1.x = acc[tt][4] + bv1.x; r1.y = acc[tt][5] + bv1.y;
    r1.z = acc[tt][6] + bv1.z; r1.w = acc[tt][7] + bv1.w;
    *reinterpret_cast<float4*>(dst + tt * 16)     = r0;
    *reinterpret_cast<float4*>(dst + tt * 16 + 4) = r1;
  }
}

// ---------------- Host-side launcher ---------------------------------------
extern "C" void kernel_launch(void* const* d_in, const int* in_sizes, int n_in,
                              void* d_out, int out_size, void* d_ws, size_t ws_size,
                              hipStream_t stream) {
  const float* x  = (const float*)d_in[0];
  const float* Wq = (const float*)d_in[1];
  const float* Wk = (const float*)d_in[2];
  const float* Wv = (const float*)d_in[3];
  const float* Wo = (const float*)d_in[4];
  const float* bo = (const float*)d_in[5];
  float* out = (float*)d_out;

  char* ws = (char*)d_ws;
  const size_t MiB = 1024 * 1024;
  bf16* Wb = (bf16*)(ws);                 // 4 x 1M bf16 =  8 MiB
  bf16* Q  = (bf16*)(ws +  8 * MiB);      // 8M bf16     = 16 MiB
  bf16* K  = (bf16*)(ws + 24 * MiB);
  bf16* Vt = (bf16*)(ws + 40 * MiB);
  bf16* O  = (bf16*)(ws + 56 * MiB);      // total 72 MiB

  k_wcvt <<<4096, 256, 0, stream>>>(Wq, Wk, Wv, Wo, Wb);
  k_qkv  <<<dim3(B_ * T_ / 16), 256, 0, stream>>>(x, Wb, Q, K, Vt);
  k_attn <<<dim3(B_ * H_, (T_ / 16) / 8), 256, 0, stream>>>(Q, K, Vt, O);
  k_oproj<<<dim3(B_ * T_ / 16, (D_ / 16) / 8), 256, 0, stream>>>(O, Wb + 3 * (size_t)(D_ * D_), bo, out);
}
